// Net_GCN_79491254714534
// MI455X (gfx1250) — matrix operand order repaired and checked by reading, not compiled, not run on hardware
//
#include <hip/hip_runtime.h>
#include <hip/hip_bf16.h>

#define HD   128   // hidden / feature width
#define GB   512   // graphs in batch
#define NCLS 10
#define EPS  1e-5f

typedef __attribute__((ext_vector_type(16))) __bf16 bf16x16;
typedef __attribute__((ext_vector_type(8)))  __bf16 bf16x8;
typedef __attribute__((ext_vector_type(8)))  float  floatx8;

__device__ __forceinline__ __bf16 f2bf(float f) {
  union { float f; unsigned u; } v; v.f = f;
  unsigned r = (v.u + 0x7FFFu + ((v.u >> 16) & 1u)) >> 16;   // RNE
  unsigned short s = (unsigned short)r;
  return __builtin_bit_cast(__bf16, s);
}

__device__ __forceinline__ void store_as(float*  y, int i, float t) { y[i] = t; }
__device__ __forceinline__ void store_as(__bf16* y, int i, float t) { y[i] = f2bf(t); }

// ---------------------------------------------------------------------------
// Degree / normalization
// ---------------------------------------------------------------------------
__global__ void k_set_ones(float* __restrict__ deg, int n) {
  int i = blockIdx.x * blockDim.x + threadIdx.x;
  if (i < n) deg[i] = 1.0f;                       // self loop contributes 1
}
__global__ void k_edge_deg(const int* __restrict__ dst, float* __restrict__ deg, int e) {
  int i = blockIdx.x * blockDim.x + threadIdx.x;
  if (i < e) atomicAdd(&deg[dst[i]], 1.0f);
}
__global__ void k_rsqrt_inplace(float* __restrict__ d, int n) {
  int i = blockIdx.x * blockDim.x + threadIdx.x;
  if (i < n) d[i] = rsqrtf(d[i]);                 // deg >= 1 always
}
__global__ void k_f32_to_bf16(const float* __restrict__ in, __bf16* __restrict__ o, int n) {
  int i = blockIdx.x * blockDim.x + threadIdx.x;
  if (i < n) o[i] = f2bf(in[i]);
}

// ---------------------------------------------------------------------------
// A-fragment loader (bf16 source): one lane needs two contiguous 8-element
// spans -> 2x global_load_b128 + register concat. Row pre-clamped (rows past
// n only feed C rows that are never stored -> exact).
// ---------------------------------------------------------------------------
__device__ __forceinline__ bf16x16 load_a_frag(const __bf16* __restrict__ A, int rowc,
                                               int k0, int half)
{
  const __bf16* base = A + (size_t)rowc * HD + k0 + half * 8;
  bf16x8 lo = ((const bf16x8*)base)[0];
  bf16x8 hi = ((const bf16x8*)(base + 16))[0];
  return __builtin_shufflevector(lo, hi, 0,1,2,3,4,5,6,7,8,9,10,11,12,13,14,15);
}

// ---------------------------------------------------------------------------
// C-tile epilogue: GUARD=false for full blocks (no exec churn), true for tail.
// ---------------------------------------------------------------------------
template <bool GUARD>
__device__ __forceinline__ void store_c(float* __restrict__ Y, const floatx8 acc0[8],
                                        const floatx8 acc1[8], const float* __restrict__ bias,
                                        int relu, int m0, int half, int lr, int n)
{
#pragma unroll
  for (int t = 0; t < 8; ++t) {
    int ncol = t * 16 + lr;
    float bv = bias ? bias[ncol] : 0.f;
#pragma unroll
    for (int j = 0; j < 8; ++j) {
      int ra = m0 + j + half * 8;
      if (!GUARD || ra < n) {
        float v = acc0[t][j] + bv;
        if (relu) v = fmaxf(v, 0.f);
        Y[(size_t)ra * HD + ncol] = v;
      }
      int rb = ra + 16;
      if (!GUARD || rb < n) {
        float v = acc1[t][j] + bv;
        if (relu) v = fmaxf(v, 0.f);
        Y[(size_t)rb * HD + ncol] = v;
      }
    }
  }
}

// ---------------------------------------------------------------------------
// WMMA GEMM:  Y[n x 128] = A_bf16[n x 128] @ W_f32[128 x 128] (+bias)(ReLU?)
// W pre-packed into LDS in per-lane B-fragment layout (2x ds_load_b128 per
// fragment). Each wave owns two 16-row tiles -> 64 v_wmma per wave.
// ---------------------------------------------------------------------------
__global__ __launch_bounds__(256) void k_gemm_wmma(
    const __bf16* __restrict__ A, const float* __restrict__ W,
    const float* __restrict__ bias, float* __restrict__ Y, int n, int relu)
{
  __shared__ bf16x16 sB[4 * 8 * 32];              // [kchunk][ntile][lane], 32 KB
  {
    __bf16* sBe = (__bf16*)sB;
    for (int wi = threadIdx.x; wi < HD * HD; wi += 256) {   // coalesced over W
      int k  = wi >> 7, nn = wi & 127;
      int kc = k >> 5, kk = k & 31;
      int hf = (kk >> 3) & 1, j = (kk & 7) | ((kk >> 4) << 3); // K within fragment
      int t  = nn >> 4, lr = nn & 15;                       // N tile / column
      sBe[((kc * 8 + t) * 32 + hf * 16 + lr) * 16 + j] = f2bf(W[wi]);
    }
  }
  __syncthreads();

  const int wave = threadIdx.x >> 5;
  const int lane = threadIdx.x & 31;
  const int half = lane >> 4;
  const int lr   = lane & 15;
  const int m0   = blockIdx.x * 256 + wave * 32;  // two 16-row tiles per wave

  floatx8 acc0[8], acc1[8];
#pragma unroll
  for (int t = 0; t < 8; ++t) {
    acc0[t] = (floatx8){0.f,0.f,0.f,0.f,0.f,0.f,0.f,0.f};
    acc1[t] = (floatx8){0.f,0.f,0.f,0.f,0.f,0.f,0.f,0.f};
  }

  const int r0c = min(m0 + lr,      n - 1);       // clamped: exact (see above)
  const int r1c = min(m0 + 16 + lr, n - 1);

#pragma unroll
  for (int kc = 0; kc < 4; ++kc) {
    const int k0 = kc * 32;
    bf16x16 a0 = load_a_frag(A, r0c, k0, half);
    bf16x16 a1 = load_a_frag(A, r1c, k0, half);
#pragma unroll
    for (int t = 0; t < 8; ++t) {
      bf16x16 b = sB[(kc * 8 + t) * 32 + lane];   // 2x ds_load_b128
      acc0[t] = __builtin_amdgcn_wmma_f32_16x16x32_bf16(
          false, a0, false, b, (short)0, acc0[t], false, false);
      acc1[t] = __builtin_amdgcn_wmma_f32_16x16x32_bf16(
          false, a1, false, b, (short)0, acc1[t], false, false);
    }
  }

  if (m0 + 32 <= n) store_c<false>(Y, acc0, acc1, bias, relu, m0, half, lr, n);
  else              store_c<true >(Y, acc0, acc1, bias, relu, m0, half, lr, n);
}

// ---------------------------------------------------------------------------
// Edge aggregation: out[dst] += dis[src]*dis[dst] * hW[src]
// one wave per message (edge or self-loop), lane handles 4 contiguous floats
// ---------------------------------------------------------------------------
__global__ __launch_bounds__(256) void k_agg(
    const float* __restrict__ hW, const int* __restrict__ src,
    const int* __restrict__ dst, const float* __restrict__ dis,
    float* __restrict__ out, int n, int e)
{
  int idx  = blockIdx.x * 8 + (threadIdx.x >> 5);
  int lane = threadIdx.x & 31;
  int total = e + n;
  if (idx >= total) return;
  int s, d; float nrm;
  if (idx < e) { s = src[idx]; d = dst[idx]; nrm = dis[s] * dis[d]; }
  else         { s = d = idx - e; float di = dis[s]; nrm = di * di; }
  float4 v = ((const float4*)(hW + (size_t)s * HD))[lane];
  float* op = out + (size_t)d * HD + lane * 4;
  atomicAdd(op + 0, v.x * nrm);
  atomicAdd(op + 1, v.y * nrm);
  atomicAdd(op + 2, v.z * nrm);
  atomicAdd(op + 3, v.w * nrm);
}

// ---------------------------------------------------------------------------
// BatchNorm: per-channel sum / sumsq reduction, then fused normalize(+ReLU).
// Output type templated: bf16 when the consumer is the next WMMA GEMM
// (identical RNE conversion the GEMM would do anyway), f32 otherwise.
// ---------------------------------------------------------------------------
__global__ void k_bn_stats(const float* __restrict__ h, float* __restrict__ stats, int n) {
  int c = threadIdx.x;                            // 128 threads = channels
  float s = 0.f, s2 = 0.f;
  for (int r = blockIdx.x; r < n; r += gridDim.x) {
    float v = h[(size_t)r * HD + c];
    s += v; s2 += v * v;
  }
  atomicAdd(&stats[c], s);
  atomicAdd(&stats[HD + c], s2);
}

template <typename T>
__global__ void k_bn_apply(const float* __restrict__ h, const float* __restrict__ stats,
                           const float* __restrict__ g, const float* __restrict__ be,
                           T* __restrict__ y, int n, int relu)
{
  int i = blockIdx.x * blockDim.x + threadIdx.x;
  if (i >= n * HD) return;
  int c = i & (HD - 1);
  float inv_n = 1.0f / (float)n;
  float m  = stats[c] * inv_n;
  float v  = stats[HD + c] * inv_n - m * m;       // biased variance
  float t  = g[c] * (h[i] - m) * rsqrtf(v + EPS) + be[c];
  store_as(y, i, relu ? fmaxf(t, 0.f) : t);
}

// ---------------------------------------------------------------------------
// Global mean pool
// ---------------------------------------------------------------------------
__global__ void k_pool_cnt(const int* __restrict__ batch, float* __restrict__ counts, int n) {
  int i = blockIdx.x * blockDim.x + threadIdx.x;
  if (i < n) atomicAdd(&counts[batch[i]], 1.0f);
}
__global__ void k_pool_sum(const float* __restrict__ h, const int* __restrict__ batch,
                           float* __restrict__ pooled, int n)
{
  int i = blockIdx.x * blockDim.x + threadIdx.x;
  if (i >= n * HD) return;
  int node = i >> 7, c = i & (HD - 1);
  atomicAdd(&pooled[(size_t)batch[node] * HD + c], h[i]);
}
__global__ void k_pool_div(float* __restrict__ pooled, const float* __restrict__ counts) {
  int i = blockIdx.x * blockDim.x + threadIdx.x;
  if (i >= GB * HD) return;
  pooled[i] /= fmaxf(counts[i >> 7], 1.0f);
}

// ---------------------------------------------------------------------------
// Head: 128 -> 10 linear, then in-place log_softmax
// ---------------------------------------------------------------------------
__global__ void k_final_linear(const float* __restrict__ z1, const float* __restrict__ lw2,
                               const float* __restrict__ lb2, float* __restrict__ out)
{
  int i = blockIdx.x * blockDim.x + threadIdx.x;
  if (i >= GB * NCLS) return;
  int g = i / NCLS, c = i % NCLS;
  float s = lb2[c];
#pragma unroll 8
  for (int k = 0; k < HD; ++k) s += z1[g * HD + k] * lw2[k * NCLS + c];
  out[i] = s;
}
__global__ void k_log_softmax(float* __restrict__ out) {
  int g = blockIdx.x * blockDim.x + threadIdx.x;
  if (g >= GB) return;
  float mx = -INFINITY;
#pragma unroll
  for (int c = 0; c < NCLS; ++c) mx = fmaxf(mx, out[g * NCLS + c]);
  float s = 0.f;
#pragma unroll
  for (int c = 0; c < NCLS; ++c) s += expf(out[g * NCLS + c] - mx);
  float l = logf(s) + mx;
#pragma unroll
  for (int c = 0; c < NCLS; ++c) out[g * NCLS + c] -= l;
}

// ---------------------------------------------------------------------------
extern "C" void kernel_launch(void* const* d_in, const int* in_sizes, int n_in,
                              void* d_out, int out_size, void* d_ws, size_t ws_size,
                              hipStream_t stream)
{
  (void)n_in; (void)out_size; (void)ws_size;
  const float* x    = (const float*)d_in[0];
  const int*   ei   = (const int*)d_in[1];
  const int*   batch= (const int*)d_in[2];
  const float* W1   = (const float*)d_in[3];   // b1/b2/b3 (idx 4,8,12) cancel in BN
  const float* g1   = (const float*)d_in[5];
  const float* be1  = (const float*)d_in[6];
  const float* W2   = (const float*)d_in[7];
  const float* g2   = (const float*)d_in[9];
  const float* be2  = (const float*)d_in[10];
  const float* W3   = (const float*)d_in[11];
  const float* g3   = (const float*)d_in[13];
  const float* be3  = (const float*)d_in[14];
  const float* gp   = (const float*)d_in[15];
  const float* bep  = (const float*)d_in[16];
  const float* lw1  = (const float*)d_in[17];
  const float* lb1  = (const float*)d_in[18];
  const float* lw2  = (const float*)d_in[19];
  const float* lb2  = (const float*)d_in[20];

  const int N = in_sizes[0] / HD;
  const int E = in_sizes[1] / 2;
  const int* srcI = ei;
  const int* dstI = ei + E;

  // workspace carve-up
  char* w = (char*)d_ws;
  size_t off = 0;
  auto take = [&](size_t bytes) {
    char* p = w + off;
    off = (off + bytes + 255) & ~(size_t)255;
    return p;
  };
  float*  dis    = (float*)take((size_t)N * 4);
  float*  bufA   = (float*)take((size_t)N * HD * 4);
  float*  bufB   = (float*)take((size_t)N * HD * 4);
  __bf16* bufC   = (__bf16*)take((size_t)N * HD * 2);   // x / h1 / h2 / zbn (bf16)
  float*  stats  = (float*)take(2 * HD * 4);
  float*  pooled = (float*)take((size_t)GB * HD * 4);
  float*  counts = (float*)take((size_t)GB * 4);
  float*  z1     = (float*)take((size_t)GB * HD * 4);
  float*  out    = (float*)d_out;

  const int TB = 256;
  const int NH = N * HD;
  const int gcnBlocks = (N + 255) / 256;          // 8 waves x 32 rows per block
  const int aggBlocks = ((E + N) + 7) / 8;

  // degree -> dis = rsqrt(deg)
  k_set_ones     <<<(N + TB - 1) / TB, TB, 0, stream>>>(dis, N);
  k_edge_deg     <<<(E + TB - 1) / TB, TB, 0, stream>>>(dstI, dis, E);
  k_rsqrt_inplace<<<(N + TB - 1) / TB, TB, 0, stream>>>(dis, N);

  // ---- layer 1: x(bf16) -> gemm -> agg -> BN(bf16) ----
  k_f32_to_bf16<<<(NH + TB - 1) / TB, TB, 0, stream>>>(x, bufC, NH);
  k_gemm_wmma<<<gcnBlocks, TB, 0, stream>>>(bufC, W1, nullptr, bufA, N, 0);
  hipMemsetAsync(bufB, 0, (size_t)NH * 4, stream);
  k_agg<<<aggBlocks, TB, 0, stream>>>(bufA, srcI, dstI, dis, bufB, N, E);
  hipMemsetAsync(stats, 0, 2 * HD * 4, stream);
  k_bn_stats<<<256, HD, 0, stream>>>(bufB, stats, N);
  k_bn_apply<__bf16><<<(NH + TB - 1) / TB, TB, 0, stream>>>(bufB, stats, g1, be1, bufC, N, 1);

  // ---- layer 2 ----
  k_gemm_wmma<<<gcnBlocks, TB, 0, stream>>>(bufC, W2, nullptr, bufA, N, 0);
  hipMemsetAsync(bufB, 0, (size_t)NH * 4, stream);
  k_agg<<<aggBlocks, TB, 0, stream>>>(bufA, srcI, dstI, dis, bufB, N, E);
  hipMemsetAsync(stats, 0, 2 * HD * 4, stream);
  k_bn_stats<<<256, HD, 0, stream>>>(bufB, stats, N);
  k_bn_apply<__bf16><<<(NH + TB - 1) / TB, TB, 0, stream>>>(bufB, stats, g2, be2, bufC, N, 1);

  // ---- layer 3 (BN output stays f32: consumed by mean-pool) ----
  k_gemm_wmma<<<gcnBlocks, TB, 0, stream>>>(bufC, W3, nullptr, bufA, N, 0);
  hipMemsetAsync(bufB, 0, (size_t)NH * 4, stream);
  k_agg<<<aggBlocks, TB, 0, stream>>>(bufA, srcI, dstI, dis, bufB, N, E);
  hipMemsetAsync(stats, 0, 2 * HD * 4, stream);
  k_bn_stats<<<256, HD, 0, stream>>>(bufB, stats, N);
  k_bn_apply<float><<<(NH + TB - 1) / TB, TB, 0, stream>>>(bufB, stats, g3, be3, bufA, N, 1);

  // ---- global mean pool ----
  hipMemsetAsync(pooled, 0, (size_t)GB * HD * 4, stream);
  hipMemsetAsync(counts, 0, (size_t)GB * 4, stream);
  k_pool_cnt<<<(N + TB - 1) / TB, TB, 0, stream>>>(batch, counts, N);
  k_pool_sum<<<(NH + TB - 1) / TB, TB, 0, stream>>>(bufA, batch, pooled, N);
  k_pool_div<<<(GB * HD + TB - 1) / TB, TB, 0, stream>>>(pooled, counts);

  // ---- BN over pooled -> bf16, FC (WMMA) + ReLU, final linear, log_softmax ----
  hipMemsetAsync(stats, 0, 2 * HD * 4, stream);
  k_bn_stats<<<256, HD, 0, stream>>>(pooled, stats, GB);
  k_bn_apply<__bf16><<<(GB * HD + TB - 1) / TB, TB, 0, stream>>>(pooled, stats, gp, bep, bufC, GB, 0);
  k_gemm_wmma<<<(GB + 255) / 256, TB, 0, stream>>>(bufC, lw1, lb1, z1, GB, 1);
  k_final_linear<<<(GB * NCLS + TB - 1) / TB, TB, 0, stream>>>(z1, lw2, lb2, out);
  k_log_softmax<<<(GB + TB - 1) / TB, TB, 0, stream>>>(out);
}
